// GAT_58171037057348
// MI455X (gfx1250) — matrix operand order repaired and checked by reading, not compiled, not run on hardware
//
#include <hip/hip_runtime.h>
#include <hip/hip_bf16.h>

// ---------------------------------------------------------------------------
// GAT (2-layer, 8-head) for MI455X / gfx1250, wave32, WMMA bf16 (f32 accum).
//
// Pipeline:
//   1. convert x -> bf16, W/W_out -> transposed bf16 [64][K] (W_out padded)
//   2. gemm_wmma:  Wh^T(bf16 [64][4096]) = A(bf16) @ B^T; the 16x512 A-tile
//      is staged into LDS by the Tensor Data Mover (tensor_load_to_lds,
//      TENSORcnt) so the inner loop only streams B from global; epilogue
//      computes src/dst = Wh @ a_{s,d} via butterfly shuffles.
//   3. gat_attn:   fused masked-softmax attention, P computed straight into
//      WMMA A-fragments, out += P @ Wh via v_wmma_f32_16x16x32_bf16,
//      row-sum rescale + ELU.  mode0: 8 heads x 1 wave -> hb bf16 [4096][512]
//      mode1: 1 head x 8 waves (j-chunks split, LDS f32-atomic reduce) -> out
//   4. log_softmax over 40 classes, in place on d_out.
// ---------------------------------------------------------------------------

typedef __attribute__((ext_vector_type(8)))  float          v8f;
typedef __attribute__((ext_vector_type(8)))  int            v8i;
typedef __attribute__((ext_vector_type(8)))  unsigned short v8u;
typedef __attribute__((ext_vector_type(16))) unsigned short v16u;
typedef __attribute__((ext_vector_type(16))) __bf16         v16bf;
typedef __attribute__((ext_vector_type(4)))  unsigned int   v4ui;
typedef __attribute__((ext_vector_type(8)))  int            v8ii;
typedef __attribute__((ext_vector_type(4)))  int            v4ii;

#define NNODES 4096
#define NFEAT  512
#define NHID   64
#define NCLASS 40
#define NHEADS 8
#define LRELU  0.2f

static __device__ __forceinline__ unsigned short f2bf(float f) {
  unsigned int u = __float_as_uint(f);
  u += 0x7FFFu + ((u >> 16) & 1u);            // round-to-nearest-even
  return (unsigned short)(u >> 16);
}

static __device__ __forceinline__ v8f wmma_bf16(v16u a, v16u b, v8f c) {
  // v_wmma_f32_16x16x32_bf16  D = A(16x32) x B(32x16) + C, f32 accumulate
  return __builtin_amdgcn_wmma_f32_16x16x32_bf16(
      false, __builtin_bit_cast(v16bf, a),
      false, __builtin_bit_cast(v16bf, b),
      (short)0, c, false, false);
}

// ---------------------------------------------------------------------------
// Tensor Data Mover: 1-D contiguous tile (2-byte elements) global -> LDS.
// D# per CDNA5 ISA ch.8: group0 = {count/flags, lds_addr, global_addr, type},
// group1 = {data_size, tensor_dim0/1, tile_dim0..2, strides}, groups 2/3 = 0
// (<=2D tensor).  Tracked by TENSORcnt.  This toolchain exposes the 6-arg
// builtin (clang-23 form): (u32x4, i32x8, i32x4, i32x4, i32x8, i32 cpol).
// ---------------------------------------------------------------------------
#if __has_builtin(__builtin_amdgcn_tensor_load_to_lds)
#define HAVE_TDM 1
static __device__ __forceinline__ void tdm_load_lds_1d(unsigned int lds_byte_off,
                                                       const void* gsrc,
                                                       unsigned int nelem2B) {
  unsigned long long ga = (unsigned long long)gsrc;
  v4ui g0 = {1u,                                   // count=1, user descriptor
             lds_byte_off,                          // lds_addr (bytes)
             (unsigned int)ga,                      // global_addr[31:0]
             (unsigned int)(ga >> 32) | (2u << 30)};// global_addr[56:32]|type=2
  v8ii g1 = {(int)(1u << 16),                       // data_size=1 -> 2 bytes
             (int)((nelem2B & 0xffffu) << 16),      // tensor_dim0[15:0]
             (int)(((nelem2B >> 16) & 0xffffu) | (1u << 16)), // dim0[31:16], tensor_dim1=1
             (int)((nelem2B & 0xffffu) << 16),      // tile_dim0 = nelem (<=65535)
             0,                                     // tile_dim1=0 (1-D), tile_dim2=0
             (int)nelem2B,                          // tensor_dim0_stride[31:0]
             0, 0};                                 // stride hi / dim1 stride
  v4ii z4 = {0, 0, 0, 0};
  v8ii z8 = {0, 0, 0, 0, 0, 0, 0, 0};
  __builtin_amdgcn_tensor_load_to_lds(g0, g1, z4, z4, z8, 0);
}
#endif

// ---------------------------------------------------------------------------
// Prep kernels: bf16 conversion + transposed (padded) weight layouts
// ---------------------------------------------------------------------------
__global__ void cvt_bf16(const float* __restrict__ in, unsigned short* __restrict__ o, int n) {
  int i = blockIdx.x * blockDim.x + threadIdx.x;
  if (i < n) o[i] = f2bf(in[i]);
}

// W [8][512][64] f32  ->  Wt [8][64][512] bf16
__global__ void prep_wtb1(const float* __restrict__ W, unsigned short* __restrict__ o) {
  int i = blockIdx.x * blockDim.x + threadIdx.x;
  if (i >= NHEADS * 64 * 512) return;
  int h = i / (64 * 512), r = i % (64 * 512);
  int n = r / 512, c = r % 512;
  o[i] = f2bf(W[(size_t)h * 512 * 64 + (size_t)c * 64 + n]);
}

// W_out [512][40] f32 -> Wt2 [64][512] bf16, rows 40..63 zero-padded
__global__ void prep_wtb2(const float* __restrict__ W, unsigned short* __restrict__ o) {
  int i = blockIdx.x * blockDim.x + threadIdx.x;
  if (i >= 64 * 512) return;
  int n = i / 512, c = i % 512;
  o[i] = (n < NCLASS) ? f2bf(W[(size_t)c * NCLASS + n]) : (unsigned short)0;
}

// ---------------------------------------------------------------------------
// GEMM: per (16-row block, head) one wave computes a 16x64 tile of Wh.
//   A[m][k]  bf16 row-major [M][K]  (x or h) -- TDM-staged into LDS
//   Bt[n][k] bf16 row-major [64][K] (pre-transposed weights) -- global stream
// Epilogue: store Wh^T bf16 [64][NNODES], and src/dst = Wh @ a_{s,d}.
// ---------------------------------------------------------------------------
__global__ void __launch_bounds__(32)
gemm_wmma(const unsigned short* __restrict__ abf,
          const unsigned short* __restrict__ btbf,
          const float* __restrict__ avec, int aStride, int adOff, int nvalid, int K,
          unsigned short* __restrict__ whbt,
          float* __restrict__ srco, float* __restrict__ dsto) {
  const int i0   = blockIdx.x * 16;
  const int head = blockIdx.y;
  const int l    = threadIdx.x;
  const int ln   = l & 15;
  const int g    = l >> 4;

  const unsigned short* bt = btbf + (size_t)head * 64 * K;
  const float* as = avec + (size_t)head * aStride;
  const float* ad = as + adOff;
  unsigned short* wh = whbt + (size_t)head * 64 * NNODES;
  float* srch = srco + (size_t)head * NNODES;
  float* dsth = dsto + (size_t)head * NNODES;

#ifdef HAVE_TDM
  // DMA the whole 16xK bf16 A-tile (contiguous 16 KB) into LDS; single wave
  // per block, so a TENSORcnt wait suffices (no barrier needed).
  __shared__ unsigned short s_a[16 * NFEAT];
  tdm_load_lds_1d((unsigned int)(size_t)(void*)s_a,
                  (const void*)(abf + (size_t)i0 * K),
                  (unsigned int)(16 * K));
  __builtin_amdgcn_s_wait_tensorcnt(0);
  const unsigned short* arow = s_a + (size_t)ln * K;
#else
  const unsigned short* arow = abf + (size_t)(i0 + ln) * K;
#endif

  v8f acc[4] = {};
  for (int kb = 0; kb < K; kb += 32) {
    // A fragment: lane (g,m=ln): K = 8g+e (e<8) and 16+8g+(e-8) (e>=8)
    v8u a0 = *(const v8u*)(arow + kb + 8 * g);
    v8u a1 = *(const v8u*)(arow + kb + 16 + 8 * g);
    v16u af = __builtin_shufflevector(a0, a1, 0, 1, 2, 3, 4, 5, 6, 7,
                                      8, 9, 10, 11, 12, 13, 14, 15);
#pragma unroll
    for (int t = 0; t < 4; ++t) {
      // B fragment: lane column n = 16t+ln, K = kb+8g+e (e<8) and kb+16+8g+(e-8) (e>=8)
      const unsigned short* brow = bt + (size_t)(t * 16 + ln) * K + kb;
      v8u b0 = *(const v8u*)(brow + 8 * g);
      v8u b1 = *(const v8u*)(brow + 16 + 8 * g);
      v16u bf = __builtin_shufflevector(b0, b1, 0, 1, 2, 3, 4, 5, 6, 7,
                                        8, 9, 10, 11, 12, 13, 14, 15);
      acc[t] = wmma_bf16(af, bf, acc[t]);
    }
  }

  // Epilogue: C/D layout — lane l, vgpr v holds (m = v+8g, n = 16t+ln)
#pragma unroll
  for (int v = 0; v < 8; ++v) {
    int m = v + 8 * g;
    float ps = 0.f, pd = 0.f;
#pragma unroll
    for (int t = 0; t < 4; ++t) {
      int n = t * 16 + ln;
      float w = acc[t][v];
      wh[(size_t)n * NNODES + i0 + m] = f2bf(w);
      float asv = (n < nvalid) ? as[n] : 0.f;
      float adv = (n < nvalid) ? ad[n] : 0.f;
      ps += w * asv;
      pd += w * adv;
    }
#pragma unroll
    for (int off = 1; off < 16; off <<= 1) {   // butterfly within 16-lane half
      ps += __shfl_xor(ps, off, 32);
      pd += __shfl_xor(pd, off, 32);
    }
    if (ln == 0) { srch[i0 + m] = ps; dsth[i0 + m] = pd; }
  }
}

// ---------------------------------------------------------------------------
// Fused GAT attention: out[i,:] = softmax_j(mask(leaky(src_i+dst_j))) @ Wh
// (no max-subtraction: logits are O(1), exp is safe; sums stay linear so
//  layer-2 splits j-chunks over 8 waves and reduces via LDS f32 atomics)
// ---------------------------------------------------------------------------
__global__ void __launch_bounds__(256)
gat_attn(const unsigned short* __restrict__ whbt,  // [heads][64][NNODES] bf16
         const float* __restrict__ src,            // [heads][NNODES]
         const float* __restrict__ dst,
         const int* __restrict__ adj,              // [NNODES][NNODES]
         int mode,                                  // 0: layer1, 1: layer2
         unsigned short* __restrict__ hb,          // mode0 out: [NNODES][512] bf16
         float* __restrict__ out) {                // mode1 out: [NNODES][40] f32
  __shared__ float s_red[16 * 64];
  __shared__ float s_sum[16];

  const int i0 = blockIdx.x * 16;
  const int w  = threadIdx.x >> 5;
  const int l  = threadIdx.x & 31;
  const int ln = l & 15;
  const int g  = l >> 4;

  const int head  = (mode == 0) ? w : 0;
  const int slice = (mode == 0) ? 0 : w;
  const int wph   = (mode == 0) ? 1 : 8;

  if (mode == 1) {
    for (int idx = threadIdx.x; idx < 16 * 64; idx += 256) s_red[idx] = 0.f;
    if (threadIdx.x < 16) s_sum[threadIdx.x] = 0.f;
    __syncthreads();
  }

  const unsigned short* wh = whbt + (size_t)head * 64 * NNODES;
  const float* dsth = dst + (size_t)head * NNODES;
  const float  srci = src[(size_t)head * NNODES + i0 + ln];
  const int* adjrow = adj + (size_t)(i0 + ln) * NNODES;

  v8f acc[4] = {};
  float rs = 0.f;

  for (int c = slice; c < NNODES / 256; c += wph) {
    const int jb = c * 256;
    if (jb + 256 * wph < NNODES)
      __builtin_prefetch(adjrow + jb + 256 * wph, 0, 1);  // global_prefetch_b8
#pragma unroll
    for (int s = 0; s < 8; ++s) {
      const int kb = jb + 32 * s;
      v8f d0 = *(const v8f*)(dsth + kb + 8 * g);
      v8f d1 = *(const v8f*)(dsth + kb + 16 + 8 * g);
      v8i m0 = *(const v8i*)(adjrow + kb + 8 * g);
      v8i m1 = *(const v8i*)(adjrow + kb + 16 + 8 * g);
      v16u af;
#pragma unroll
      for (int e = 0; e < 8; ++e) {
        float v0 = srci + d0[e];
        v0 = (v0 > 0.f) ? v0 : LRELU * v0;
        float p0 = (m0[e] > 0) ? __expf(v0) : 0.f;
        float v1 = srci + d1[e];
        v1 = (v1 > 0.f) ? v1 : LRELU * v1;
        float p1 = (m1[e] > 0) ? __expf(v1) : 0.f;
        rs += p0 + p1;
        af[e]     = f2bf(p0);
        af[e + 8] = f2bf(p1);
      }
#pragma unroll
      for (int t = 0; t < 4; ++t) {
        const unsigned short* brow = wh + (size_t)(t * 16 + ln) * NNODES + kb;
        v8u b0 = *(const v8u*)(brow + 8 * g);
        v8u b1 = *(const v8u*)(brow + 16 + 8 * g);
        v16u bf = __builtin_shufflevector(b0, b1, 0, 1, 2, 3, 4, 5, 6, 7,
                                          8, 9, 10, 11, 12, 13, 14, 15);
        acc[t] = wmma_bf16(af, bf, acc[t]);
      }
    }
  }

  if (mode == 0) {
    float rtot = rs + __shfl_xor(rs, 16, 32);     // combine g=0/1 partials
#pragma unroll
    for (int v = 0; v < 8; ++v) {
      int m = v + 8 * g;
      float inv = 1.0f / __shfl(rtot, m, 32);     // row sum lives in lane m
#pragma unroll
      for (int t = 0; t < 4; ++t) {
        float o = acc[t][v] * inv;
        o = (o > 0.f) ? o : (__expf(o) - 1.0f);   // ELU per head
        hb[(size_t)(i0 + m) * (NHEADS * NHID) + head * NHID + t * 16 + ln] = f2bf(o);
      }
    }
  } else {
#pragma unroll
    for (int v = 0; v < 8; ++v) {
      int m = v + 8 * g;
#pragma unroll
      for (int t = 0; t < 4; ++t)
        atomicAdd(&s_red[m * 64 + t * 16 + ln], acc[t][v]);  // ds_add_f32
    }
    atomicAdd(&s_sum[ln], rs);
    __syncthreads();
    for (int idx = threadIdx.x; idx < 16 * NCLASS; idx += 256) {
      int m = idx / NCLASS, n = idx % NCLASS;
      float o = s_red[m * 64 + n] / s_sum[m];
      o = (o > 0.f) ? o : (__expf(o) - 1.0f);     // ELU on output layer
      out[(size_t)(i0 + m) * NCLASS + n] = o;
    }
  }
}

__global__ void logsoftmax_rows(float* __restrict__ out) {
  int r = blockIdx.x * blockDim.x + threadIdx.x;
  if (r >= NNODES) return;
  float* row = out + (size_t)r * NCLASS;
  float m = row[0];
  for (int n = 1; n < NCLASS; ++n) m = fmaxf(m, row[n]);
  float s = 0.f;
  for (int n = 0; n < NCLASS; ++n) s += __expf(row[n] - m);
  float lse = m + __logf(s);
  for (int n = 0; n < NCLASS; ++n) row[n] -= lse;
}

// ---------------------------------------------------------------------------
extern "C" void kernel_launch(void* const* d_in, const int* in_sizes, int n_in,
                              void* d_out, int out_size, void* d_ws, size_t ws_size,
                              hipStream_t stream) {
  (void)in_sizes; (void)n_in; (void)out_size; (void)ws_size;
  const float* x     = (const float*)d_in[0];
  const int*   adj   = (const int*)d_in[1];
  const float* W     = (const float*)d_in[2];
  const float* a     = (const float*)d_in[3];
  const float* W_out = (const float*)d_in[4];
  const float* a_out = (const float*)d_in[5];
  float* out = (float*)d_out;

  char* p = (char*)d_ws;
  auto alloc = [&](size_t bytes) {
    char* q = p;
    p += (bytes + 255) & ~(size_t)255;
    return q;
  };
  unsigned short* xb    = (unsigned short*)alloc((size_t)NNODES * NFEAT * 2);
  unsigned short* hb    = (unsigned short*)alloc((size_t)NNODES * NFEAT * 2);
  unsigned short* wtb1  = (unsigned short*)alloc((size_t)NHEADS * 64 * 512 * 2);
  unsigned short* wtb2  = (unsigned short*)alloc((size_t)64 * 512 * 2);
  unsigned short* whbt1 = (unsigned short*)alloc((size_t)NHEADS * 64 * NNODES * 2);
  unsigned short* whbt2 = (unsigned short*)alloc((size_t)64 * NNODES * 2);
  float* src1 = (float*)alloc((size_t)NHEADS * NNODES * 4);
  float* dst1 = (float*)alloc((size_t)NHEADS * NNODES * 4);
  float* src2 = (float*)alloc((size_t)NNODES * 4);
  float* dst2 = (float*)alloc((size_t)NNODES * 4);

  cvt_bf16<<<(NNODES * NFEAT + 255) / 256, 256, 0, stream>>>(x, xb, NNODES * NFEAT);
  prep_wtb1<<<(NHEADS * 64 * 512 + 255) / 256, 256, 0, stream>>>(W, wtb1);
  prep_wtb2<<<(64 * 512 + 255) / 256, 256, 0, stream>>>(W_out, wtb2);

  // Layer 1: Wh_k = x @ W_k (+ src/dst), then fused attention -> hb
  gemm_wmma<<<dim3(NNODES / 16, NHEADS), 32, 0, stream>>>(
      xb, wtb1, a, 2 * NHID, NHID, NHID, NFEAT, whbt1, src1, dst1);
  gat_attn<<<NNODES / 16, 256, 0, stream>>>(whbt1, src1, dst1, adj, 0, hb, nullptr);

  // Layer 2: Wh2 = h @ W_out (padded to 64 cols), attention -> ELU -> out
  gemm_wmma<<<dim3(NNODES / 16, 1), 32, 0, stream>>>(
      hb, wtb2, a_out, 0, NCLASS, NCLASS, NFEAT, whbt2, src2, dst2);
  gat_attn<<<NNODES / 16, 256, 0, stream>>>(whbt2, src2, dst2, adj, 1, nullptr, out);

  logsoftmax_rows<<<(NNODES + 255) / 256, 256, 0, stream>>>(out);
}
